// PATvidx4_51642686767657
// MI455X (gfx1250) — matrix operand order repaired and checked by reading, not compiled, not run on hardware
//
#include <hip/hip_runtime.h>
#include <hip/hip_bf16.h>

// ---------------------------------------------------------------------------
// Types for WMMA
// ---------------------------------------------------------------------------
typedef __bf16 bf16_t;
typedef __attribute__((ext_vector_type(16))) __bf16 v16bf;
typedef __attribute__((ext_vector_type(8)))  float  v8f;

#define CONV_WAVES 4   // waves per block; each wave owns a 32(co) x 16(pix) tile

// ---------------------------------------------------------------------------
// Implicit-GEMM convolution via V_WMMA_F32_16X16X32_BF16.
//   GEMM view: D[co][p] = sum_{tap,ci} W[co][ci][tap] * x[ci][pix(tap,p)]
//   K-order is tap-major / channel-minor; OIHW weights are read at stride K*K.
//   Every CI here is 1 or a multiple of 32, so each 32-wide K chunk sits in a
//   single tap: (fy,fx) address math hoists out of the element loop entirely.
//   M tile = 32 output channels (two 16x16x32 WMMAs share one B fragment),
//   N tile = 16 pixels per wave. A tile staged via LDS, shared by 4 waves.
// Fused: bias, leaky-relu(0.1), residual add, channel-offset input/output.
// K is the (square) filter size, compile-time -> tap loop fully unrolls.
// ---------------------------------------------------------------------------
template<int K>
__global__ __launch_bounds__(32 * CONV_WAVES)
void conv_wmma_kernel(const float* __restrict__ x,
                      const float* __restrict__ w,
                      const float* __restrict__ bias,
                      const float* __restrict__ res,
                      float* __restrict__ out,
                      int Nb, int CI, int H, int W, int CO,
                      int stride, int dil, int pad,
                      int OH, int OW,
                      int CI_total, int ci_off,
                      int CO_total, int co_off,
                      int act)
{
    __shared__ bf16_t Alds[32 * 32];          // A tile: [m][ci] row-major, 2 KB

    const int lane = threadIdx.x & 31;
    const int wv   = threadIdx.x >> 5;
    const int l16  = lane & 15;
    const int half = lane >> 4;

    const int co0 = blockIdx.y * 32;
    const int P   = Nb * OH * OW;
    const int p   = (blockIdx.x * CONV_WAVES + wv) * 16 + l16;   // B column
    const bool pvalid = p < P;
    const int pc  = pvalid ? p : 0;

    const int nb  = pc / (OH * OW);
    const int rem = pc % (OH * OW);
    const int oh  = rem / OW;
    const int ow  = rem % OW;
    const int ih0 = oh * stride - pad;
    const int iw0 = ow * stride - pad;

    const int HWin = H * W;
    const float* xbase = x + (size_t)(nb * CI_total + ci_off) * HWin;

    v8f acc0 = {};
    v8f acc1 = {};

#pragma unroll
    for (int tap = 0; tap < K * K; ++tap) {
        const int fy = tap / K;
        const int fx = tap % K;
        const int ih = ih0 + fy * dil;
        const int iw = iw0 + fx * dil;
        const bool vpix = pvalid &&
                          (unsigned)ih < (unsigned)H &&
                          (unsigned)iw < (unsigned)W;
        const float* bp = xbase + (size_t)ih * W + iw;   // + ci*HWin per elem

        for (int cb = 0; cb < CI; cb += 32) {
            __syncthreads();                  // protect previous tile read
            // ---- stage A tile: rows co0..co0+31, cols ci = cb..cb+31 ----
            for (int t = threadIdx.x; t < 1024; t += 32 * CONV_WAVES) {
                int m  = t >> 5;
                int k  = t & 31;
                int co = co0 + m;
                int ci = cb + k;
                float v = 0.0f;
                if (co < CO && ci < CI)
                    v = w[((size_t)co * CI + ci) * (K * K) + tap];
                Alds[t] = (bf16_t)v;
            }
            __syncthreads();

            // ---- two A fragments (ISA 16-bit A per-lane K interleave) ----
            v16bf a0, a1;
#pragma unroll
            for (int e = 0; e < 16; ++e) {
                int v   = e >> 1;
                int pos = e & 1;
                int kin = (v < 4) ? (half * 8 + v * 2 + pos)
                                  : (16 + half * 8 + (v - 4) * 2 + pos);
                a0[e] = Alds[l16 * 32 + kin];
                a1[e] = Alds[(l16 + 16) * 32 + kin];
            }

            // ---- B fragment: lane pixel = N col, K elem = same interleave as A ----
            v16bf bfr;
#pragma unroll
            for (int e = 0; e < 16; ++e) {
                int ci = cb + ((e < 8) ? (half * 8 + e) : (16 + half * 8 + (e - 8)));
                float v = (vpix && ci < CI) ? bp[(size_t)ci * HWin] : 0.0f;
                bfr[e] = (bf16_t)v;
            }

            acc0 = __builtin_amdgcn_wmma_f32_16x16x32_bf16(
                       false, a0, false, bfr, (short)0, acc0, false, false);
            acc1 = __builtin_amdgcn_wmma_f32_16x16x32_bf16(
                       false, a1, false, bfr, (short)0, acc1, false, false);
        }
    }

    // ---- store D: acc0 rows co0+half*8+v, acc1 rows +16; col = pixel p ----
    if (pvalid) {
        const size_t opix = (size_t)oh * OW + ow;
#pragma unroll
        for (int v = 0; v < 8; ++v) {
            int co = co0 + half * 8 + v;
            if (co < CO) {
                float r = acc0[v];
                if (bias) r += bias[co];
                if (act)  r = (r > 0.0f) ? r : 0.1f * r;
                if (res)  r += res[((size_t)nb * CO + co) * OH * OW + opix];
                out[((size_t)nb * CO_total + co_off + co) * OH * OW + opix] = r;
            }
            int co2 = co + 16;
            if (co2 < CO) {
                float r = acc1[v];
                if (bias) r += bias[co2];
                if (act)  r = (r > 0.0f) ? r : 0.1f * r;
                if (res)  r += res[((size_t)nb * CO + co2) * OH * OW + opix];
                out[((size_t)nb * CO_total + co_off + co2) * OH * OW + opix] = r;
            }
        }
    }
}

// ---------------------------------------------------------------------------
// Elementwise add:  o = a + b
// ---------------------------------------------------------------------------
__global__ void add2_kernel(float* __restrict__ o, const float* __restrict__ a,
                            const float* __restrict__ b, int n)
{
    int i = blockIdx.x * blockDim.x + threadIdx.x;
    if (i < n) o[i] = a[i] + b[i];
}

// ---------------------------------------------------------------------------
// Pixel shuffle r=4:  (N, Cout*r*r, H, W) -> (N, Cout, H*r, W*r)
// ---------------------------------------------------------------------------
__global__ void pixel_shuffle_kernel(const float* __restrict__ in,
                                     float* __restrict__ out,
                                     int N, int Cout, int H, int W, int r)
{
    int i = blockIdx.x * blockDim.x + threadIdx.x;
    int wR = W * r, hR = H * r;
    int total = N * Cout * hR * wR;
    if (i >= total) return;
    int ow = i % wR; int t = i / wR;
    int oh = t % hR; t /= hR;
    int c  = t % Cout; int n = t / Cout;
    int h = oh / r, i1 = oh % r;
    int ww = ow / r, j1 = ow % r;
    int cin = c * r * r + i1 * r + j1;
    out[i] = in[((n * (Cout * r * r) + cin) * H + h) * W + ww];
}

// ---------------------------------------------------------------------------
// PAM gather-attention. One block (64 thr) per (batch, query-pixel).
// ---------------------------------------------------------------------------
__global__ __launch_bounds__(64)
void pam_attn_kernel(const float* __restrict__ Q,
                     const float* __restrict__ S,
                     const float* __restrict__ Rv,
                     const int* __restrict__ pos,
                     float* __restrict__ outc,
                     float* __restrict__ Mout,
                     int Cv, int C_total, int c_off)
{
    const int HW = 48 * 48;
    int q = blockIdx.x % HW;
    int b = blockIdx.x / HW;

    __shared__ float qv[32];
    __shared__ float sc[49];
    __shared__ int pxs[49], pys[49];

    int tid = threadIdx.x;
    if (tid < 32) qv[tid] = Q[(b * 32 + tid) * HW + q];
    if (tid < 49) {
        pxs[tid] = pos[((b * 2 + 0) * HW + q) * 49 + tid];
        pys[tid] = pos[((b * 2 + 1) * HW + q) * 49 + tid];
    }
    __syncthreads();

    if (tid < 49) {
        const float* sp = S + (size_t)(b * 32) * HW + pxs[tid] * 48 + pys[tid];
        float s = 0.0f;
#pragma unroll 8
        for (int c = 0; c < 32; ++c) s += qv[c] * sp[c * HW];
        sc[tid] = s;
    }
    __syncthreads();

    if (tid == 0) {
        float m = sc[0];
        for (int j = 1; j < 49; ++j) m = fmaxf(m, sc[j]);
        float sum = 0.0f;
        for (int j = 0; j < 49; ++j) { float e = __expf(sc[j] - m); sc[j] = e; sum += e; }
        float inv = 1.0f / sum;
        for (int j = 0; j < 49; ++j) sc[j] *= inv;
    }
    __syncthreads();

    if (tid < 49) Mout[((size_t)b * HW + q) * 49 + tid] = sc[tid];

    for (int v = tid; v < Cv; v += 64) {
        const float* rp = Rv + (size_t)(b * Cv + v) * HW;
        float o = 0.0f;
        for (int j = 0; j < 49; ++j) o += sc[j] * rp[pxs[j] * 48 + pys[j]];
        outc[((size_t)b * C_total + c_off + v) * HW + q] = o;
    }
}

// ---------------------------------------------------------------------------
// Host-side helpers
// ---------------------------------------------------------------------------
struct RB  { const float *w1, *w2; };
struct ASP { const float *c1[3], *c2[3], *c3[3], *b1, *b2, *b3; };

static void launch_conv(hipStream_t s,
                        const float* x, const float* w, const float* bias,
                        const float* res, float* out,
                        int Nb, int CI, int H, int W, int CO, int K,
                        int stride, int dil,
                        int CI_total, int ci_off, int CO_total, int co_off, int act)
{
    int pad = ((K - 1) / 2) * dil;
    int eff = (K - 1) * dil + 1;
    int OH = (H + 2 * pad - eff) / stride + 1;
    int OW = (W + 2 * pad - eff) / stride + 1;
    int P = Nb * OH * OW;
    dim3 grid((P + 16 * CONV_WAVES - 1) / (16 * CONV_WAVES), (CO + 31) / 32);
    if (K == 1) {
        conv_wmma_kernel<1><<<grid, 32 * CONV_WAVES, 0, s>>>(
            x, w, bias, res, out, Nb, CI, H, W, CO, stride, dil, pad, OH, OW,
            CI_total, ci_off, CO_total, co_off, act);
    } else {
        conv_wmma_kernel<3><<<grid, 32 * CONV_WAVES, 0, s>>>(
            x, w, bias, res, out, Nb, CI, H, W, CO, stride, dil, pad, OH, OW,
            CI_total, ci_off, CO_total, co_off, act);
    }
}

static void resb_run(hipStream_t s, const float* x, float* tmp, float* out,
                     const RB& rb, int Nb, int C, int H, int W,
                     int CO_total, int co_off)
{
    launch_conv(s, x, rb.w1, nullptr, nullptr, tmp, Nb, C, H, W, C, 3, 1, 1,
                C, 0, C, 0, 1);
    launch_conv(s, tmp, rb.w2, nullptr, x, out, Nb, C, H, W, C, 3, 1, 1,
                C, 0, CO_total, co_off, 0);
}

static void branch_run(hipStream_t s, const float* y, float* BR,
                       const float* const* c, int Nb, int C, int H, int W)
{
    launch_conv(s, y, c[0], nullptr, nullptr, BR, Nb, C, H, W, C, 3, 1, 1,
                C, 0, 3 * C, 0, 1);
    launch_conv(s, y, c[1], nullptr, nullptr, BR, Nb, C, H, W, C, 3, 1, 4,
                C, 0, 3 * C, C, 1);
    launch_conv(s, y, c[2], nullptr, nullptr, BR, Nb, C, H, W, C, 3, 1, 8,
                C, 0, 3 * C, 2 * C, 1);
}

static void aspp_run(hipStream_t s, const float* x, float* out,
                     float* BR, float* B1, float* B2b, float* ACC,
                     const ASP& a, int Nb, int C, int H, int W)
{
    int C3 = 3 * C;
    int n = Nb * C * H * W;
    dim3 ag((n + 255) / 256);

    branch_run(s, x, BR, a.c1, Nb, C, H, W);
    launch_conv(s, BR, a.b1, nullptr, nullptr, B1, Nb, C3, H, W, C, 1, 1, 1,
                C3, 0, C, 0, 0);
    add2_kernel<<<ag, 256, 0, s>>>(ACC, x, B1, n);

    branch_run(s, B1, BR, a.c2, Nb, C, H, W);
    launch_conv(s, BR, a.b2, nullptr, nullptr, B2b, Nb, C3, H, W, C, 1, 1, 1,
                C3, 0, C, 0, 0);
    add2_kernel<<<ag, 256, 0, s>>>(ACC, ACC, B2b, n);

    branch_run(s, B2b, BR, a.c3, Nb, C, H, W);
    launch_conv(s, BR, a.b3, nullptr, ACC, out, Nb, C3, H, W, C, 1, 1, 1,
                C3, 0, C, 0, 0);
}

// ---------------------------------------------------------------------------
// kernel_launch: full network
// ---------------------------------------------------------------------------
extern "C" void kernel_launch(void* const* d_in, const int* in_sizes, int n_in,
                              void* d_out, int out_size, void* d_ws, size_t ws_size,
                              hipStream_t stream)
{
    (void)in_sizes; (void)n_in; (void)out_size; (void)ws_size;

    const float* lr0   = (const float*)d_in[0];
    const float* lr1   = (const float*)d_in[1];
    const float* lr2   = (const float*)d_in[2];
    const float* hr_in = (const float*)d_in[3];
    const int*   Pos   = (const int*)d_in[4];
    // d_in[5] = is_training (0 in setup; full tuple output is expected)

    int wi = 6;
    auto nw = [&]() { return (const float*)d_in[wi++]; };
    auto rdRB  = [&](RB& r)  { r.w1 = nw(); r.w2 = nw(); };
    auto rdASP = [&](ASP& a) {
        for (int i = 0; i < 3; ++i) a.c1[i] = nw();
        for (int i = 0; i < 3; ++i) a.c2[i] = nw();
        for (int i = 0; i < 3; ++i) a.c3[i] = nw();
        a.b1 = nw(); a.b2 = nw(); a.b3 = nw();
    };

    // --- params, flattened in dict-insertion order ---
    const float* lf_w0 = nw(); RB lf_rb1; rdRB(lf_rb1); ASP lf_a1; rdASP(lf_a1);
    RB lf_rb2; rdRB(lf_rb2);   ASP lf_a2; rdASP(lf_a2); RB lf_rb3; rdRB(lf_rb3);

    const float* hf_w0 = nw(); RB hf_rb1; rdRB(hf_rb1); ASP hf_a1; rdASP(hf_a1);
    RB hf_rb2; rdRB(hf_rb2);   const float* hf_d1 = nw();
    RB hf_rb3; rdRB(hf_rb3);   ASP hf_a2; rdASP(hf_a2); RB hf_rb4; rdRB(hf_rb4);
    const float* hf_d2 = nw();
    RB hf_rb5; rdRB(hf_rb5);   ASP hf_a3; rdASP(hf_a3); RB hf_rb6; rdRB(hf_rb6);

    RB pm_lr_rb; rdRB(pm_lr_rb); RB pm_hr_rb; rdRB(pm_hr_rb);
    const float* q_w = nw();  const float* q_b = nw();
    const float* k_w = nw();  const float* k_b = nw();
    const float* hk_w = nw(); const float* hk_b = nw();
    const float* f_w = nw();  const float* f_b = nw();

    RB up_rb1; rdRB(up_rb1); RB up_rb2; rdRB(up_rb2);
    RB up_rb3; rdRB(up_rb3); RB up_rb4; rdRB(up_rb4);
    const float* wps = nw();
    RB up_rb5; rdRB(up_rb5); RB up_rb6; rdRB(up_rb6);
    const float* wout = nw();

    // --- workspace layout (floats) ---
    float* ws = (float*)d_ws;
    size_t off = 0;
    auto alloc = [&](size_t n) { float* p = ws + off; off += n; return p; };
    const size_t HRA = (size_t)2 * 64 * 192 * 192;     // largest activation
    float* bufA = alloc(HRA);
    float* bufB = alloc(HRA);
    float* T    = alloc(HRA);
    float* B2b  = alloc(HRA);
    float* ACC  = alloc(HRA);
    float* BR   = alloc((size_t)2 * 192 * 192 * 192);  // 3C branch concat
    float* LRb0 = alloc((size_t)2 * 64 * 48 * 48);
    float* LRb1 = alloc((size_t)2 * 64 * 48 * 48);
    float* LRb2 = alloc((size_t)2 * 64 * 48 * 48);
    float* HRBUF = alloc((size_t)2 * 128 * 48 * 48);
    float* LRB1r = alloc((size_t)2 * 64 * 48 * 48);
    float* LRB2r = alloc((size_t)2 * 64 * 48 * 48);
    float* HRBr  = alloc((size_t)2 * 128 * 48 * 48);
    float* QB = alloc((size_t)2 * 32 * 48 * 48);
    float* S1 = alloc((size_t)2 * 32 * 48 * 48);
    float* S2 = alloc((size_t)2 * 32 * 48 * 48);
    float* S3 = alloc((size_t)2 * 32 * 48 * 48);
    float* CONCAT = alloc((size_t)2 * 320 * 48 * 48);
    float* FUSED  = alloc((size_t)2 * 128 * 48 * 48);

    hipStream_t s = stream;

    // ================= lr_feature (x3 inputs, shared weights) ================
    const float* lrin[3] = { lr0, lr1, lr2 };
    float* lrout[3] = { LRb0, LRb1, LRb2 };
    for (int i = 0; i < 3; ++i) {
        launch_conv(s, lrin[i], lf_w0, nullptr, nullptr, bufA,
                    2, 1, 48, 48, 64, 3, 1, 1, 1, 0, 64, 0, 1);
        resb_run(s, bufA, T, bufB, lf_rb1, 2, 64, 48, 48, 64, 0);
        aspp_run(s, bufB, bufA, BR, T, B2b, ACC, lf_a1, 2, 64, 48, 48);
        resb_run(s, bufA, T, bufB, lf_rb2, 2, 64, 48, 48, 64, 0);
        aspp_run(s, bufB, bufA, BR, T, B2b, ACC, lf_a2, 2, 64, 48, 48);
        resb_run(s, bufA, T, lrout[i], lf_rb3, 2, 64, 48, 48, 64, 0);
    }

    // ================= hr_feature ================
    launch_conv(s, hr_in, hf_w0, nullptr, nullptr, bufA,
                2, 1, 192, 192, 64, 3, 1, 1, 1, 0, 64, 0, 1);
    resb_run(s, bufA, T, bufB, hf_rb1, 2, 64, 192, 192, 64, 0);
    aspp_run(s, bufB, bufA, BR, T, B2b, ACC, hf_a1, 2, 64, 192, 192);
    resb_run(s, bufA, T, bufB, hf_rb2, 2, 64, 192, 192, 64, 0);
    launch_conv(s, bufB, hf_d1, nullptr, nullptr, bufA,
                2, 64, 192, 192, 96, 3, 2, 1, 64, 0, 96, 0, 0);
    resb_run(s, bufA, T, bufB, hf_rb3, 2, 96, 96, 96, 96, 0);
    aspp_run(s, bufB, bufA, BR, T, B2b, ACC, hf_a2, 2, 96, 96, 96);
    resb_run(s, bufA, T, bufB, hf_rb4, 2, 96, 96, 96, 96, 0);
    launch_conv(s, bufB, hf_d2, nullptr, nullptr, bufA,
                2, 96, 96, 96, 128, 3, 2, 1, 96, 0, 128, 0, 0);
    resb_run(s, bufA, T, bufB, hf_rb5, 2, 128, 48, 48, 128, 0);
    aspp_run(s, bufB, bufA, BR, T, B2b, ACC, hf_a3, 2, 128, 48, 48);
    resb_run(s, bufA, T, HRBUF, hf_rb6, 2, 128, 48, 48, 128, 0);

    // ================= pam ================
    resb_run(s, LRb0, T, CONCAT, pm_lr_rb, 2, 64, 48, 48, 320, 256);
    resb_run(s, LRb1, T, LRB1r, pm_lr_rb, 2, 64, 48, 48, 64, 0);
    resb_run(s, LRb2, T, LRB2r, pm_lr_rb, 2, 64, 48, 48, 64, 0);
    resb_run(s, HRBUF, T, HRBr, pm_hr_rb, 2, 128, 48, 48, 128, 0);

    launch_conv(s, CONCAT, q_w, q_b, nullptr, QB,
                2, 64, 48, 48, 32, 1, 1, 1, 320, 256, 32, 0, 0);
    launch_conv(s, LRB1r, k_w, k_b, nullptr, S1,
                2, 64, 48, 48, 32, 1, 1, 1, 64, 0, 32, 0, 0);
    launch_conv(s, LRB2r, k_w, k_b, nullptr, S2,
                2, 64, 48, 48, 32, 1, 1, 1, 64, 0, 32, 0, 0);
    launch_conv(s, HRBr, hk_w, hk_b, nullptr, S3,
                2, 128, 48, 48, 32, 1, 1, 1, 128, 0, 32, 0, 0);

    const int HW = 48 * 48;
    const size_t psz = (size_t)2 * 2 * HW * 49;        // per-branch Pos size
    float* Mbase = (float*)d_out + (size_t)2 * 3 * 192 * 192;
    const size_t Msz = (size_t)2 * HW * 49;

    pam_attn_kernel<<<2 * HW, 64, 0, s>>>(QB, S1, LRb1, Pos + 0 * psz,
                                          CONCAT, Mbase + 0 * Msz, 64, 320, 0);
    pam_attn_kernel<<<2 * HW, 64, 0, s>>>(QB, S2, LRb2, Pos + 1 * psz,
                                          CONCAT, Mbase + 1 * Msz, 64, 320, 64);
    pam_attn_kernel<<<2 * HW, 64, 0, s>>>(QB, S3, HRBUF, Pos + 2 * psz,
                                          CONCAT, Mbase + 2 * Msz, 128, 320, 128);

    launch_conv(s, CONCAT, f_w, f_b, nullptr, FUSED,
                2, 320, 48, 48, 128, 1, 1, 1, 320, 0, 128, 0, 0);

    // ================= upscale ================
    resb_run(s, FUSED, T, bufA, up_rb1, 2, 128, 48, 48, 128, 0);
    resb_run(s, bufA, T, bufB, up_rb2, 2, 128, 48, 48, 128, 0);
    resb_run(s, bufB, T, bufA, up_rb3, 2, 128, 48, 48, 128, 0);
    resb_run(s, bufA, T, bufB, up_rb4, 2, 128, 48, 48, 128, 0);
    launch_conv(s, bufB, wps, nullptr, nullptr, bufA,
                2, 128, 48, 48, 512, 1, 1, 1, 128, 0, 512, 0, 0);
    {
        int total = 2 * 32 * 192 * 192;
        pixel_shuffle_kernel<<<(total + 255) / 256, 256, 0, s>>>(
            bufA, bufB, 2, 32, 48, 48, 4);
    }
    resb_run(s, bufB, T, bufA, up_rb5, 2, 32, 192, 192, 32, 0);
    resb_run(s, bufA, T, bufB, up_rb6, 2, 32, 192, 192, 32, 0);
    launch_conv(s, bufB, wout, nullptr, nullptr, (float*)d_out,
                2, 32, 192, 192, 3, 3, 1, 1, 32, 0, 3, 0, 0);
}
